// SpiralAttention_65678639891071
// MI455X (gfx1250) — compile-verified, matrix operand order checked by reading, not run on hardware
//
#include <hip/hip_runtime.h>

typedef __attribute__((ext_vector_type(16))) __bf16 v16bf;
typedef __attribute__((ext_vector_type(8)))  float  v8f;
typedef __attribute__((ext_vector_type(2)))  __bf16 bf16x2;
typedef __attribute__((ext_vector_type(4)))  __bf16 bf16x4;
typedef __attribute__((ext_vector_type(2)))  float  f32x2;
typedef __attribute__((ext_vector_type(4)))  float  f32x4;

__device__ __forceinline__ bf16x2 cvt_pk_bf16(float x, float y) {
    f32x2 f = {x, y};
    return __builtin_convertvector(f, bf16x2);      // -> v_cvt_pk_bf16_f32
}
__device__ __forceinline__ bf16x4 cvt_bf16x4(float4 f) {
    f32x4 v = {f.x, f.y, f.z, f.w};
    return __builtin_convertvector(v, bf16x4);
}

// ---------------------------------------------------------------------------
// C[M,N] = A[M,K] @ W[K,N] + bias[N], bf16 WMMA, fp32 accumulate.
// WG tile 128x128, 8 waves, each wave 32x64 (2x4 WMMA tiles).
// K-loop software-pipelined with ping-pong LDS double buffering:
// one barrier per iteration; next tile's cvt+ds_store overlaps WMMAs.
// ---------------------------------------------------------------------------
__global__ __launch_bounds__(256) void gemm_bias_bf16(
    const float* __restrict__ A, const float* __restrict__ W,
    const float* __restrict__ bias, float* __restrict__ C,
    int M, int N, int K, int lda, int ldb, int ldc)
{
    __shared__ __align__(16) __bf16 lds_a[2][128 * 32];
    __shared__ __align__(16) __bf16 lds_b[2][128 * 32];

    const int tid    = threadIdx.x;
    const int lane   = tid & 31;
    const int wave   = tid >> 5;
    const int wr     = wave >> 1;        // 0..3 (M dir)
    const int wc     = wave & 1;         // 0..1 (N dir)
    const int lane15 = lane & 15;
    const int half   = lane >> 4;        // 0 or 1
    const int mBlock = blockIdx.y * 128;
    const int nBlock = blockIdx.x * 128;

    // per-thread staging coordinates (4 float4 each for A and W tiles)
    int arow[4], akq[4], brow[4], bnq[4];
#pragma unroll
    for (int l = 0; l < 4; ++l) {
        int idx = tid + l * 256;
        arow[l] = idx >> 3;            // 0..127
        akq[l]  = (idx & 7) << 2;      // 0..28
        brow[l] = idx >> 5;            // 0..31
        bnq[l]  = (idx & 31) << 2;     // 0..124
    }

    v8f acc[2][4] = {};
    float4 ra[4], rb[4];

    auto stage_loads = [&](int kk) {
#pragma unroll
        for (int l = 0; l < 4; ++l)
            ra[l] = *(const float4*)(A + (size_t)(mBlock + arow[l]) * lda + kk + akq[l]);
#pragma unroll
        for (int l = 0; l < 4; ++l)
            rb[l] = *(const float4*)(W + (size_t)(kk + brow[l]) * ldb + nBlock + bnq[l]);
    };
    auto commit = [&](int buf) {
        __bf16* la = &lds_a[buf][0];
        __bf16* lb = &lds_b[buf][0];
#pragma unroll
        for (int l = 0; l < 4; ++l)
            *(bf16x4*)(&la[arow[l] * 32 + akq[l]]) = cvt_bf16x4(ra[l]);
#pragma unroll
        for (int l = 0; l < 4; ++l) {
            lb[(bnq[l] + 0) * 32 + brow[l]] = (__bf16)rb[l].x;
            lb[(bnq[l] + 1) * 32 + brow[l]] = (__bf16)rb[l].y;
            lb[(bnq[l] + 2) * 32 + brow[l]] = (__bf16)rb[l].z;
            lb[(bnq[l] + 3) * 32 + brow[l]] = (__bf16)rb[l].w;
        }
    };

    // prologue: stage + commit first K tile into buffer 0
    stage_loads(0);
    commit(0);

    int p = 0;
    for (int k0 = 0; k0 < K; k0 += 32) {
        __syncthreads();                       // buf[p] visible to all waves
        const bool hasNext = (k0 + 32 < K);
        if (hasNext) {
            stage_loads(k0 + 32);              // global loads fly under WMMAs
            if (k0 + 64 < K)
                __builtin_prefetch(A + (size_t)(mBlock + (tid >> 1)) * lda + k0 + 64, 0, 1);
        }

        // ---- fragments + WMMA from buf[p] ----
        const int kb = half * 8;   // lane-half K offset per ISA 16-bit layout
        const __bf16* la = &lds_a[p][0];
        const __bf16* lb = &lds_b[p][0];
        union { v16bf v; uint4 q[2]; } afrag[2], bfrag[4];
#pragma unroll
        for (int mi = 0; mi < 2; ++mi) {
            int row = wr * 32 + mi * 16 + lane15;
            afrag[mi].q[0] = *(const uint4*)(&la[row * 32 + kb]);
            afrag[mi].q[1] = *(const uint4*)(&la[row * 32 + kb + 16]);
        }
#pragma unroll
        for (int ni = 0; ni < 4; ++ni) {
            int col = wc * 64 + ni * 16 + lane15;
            bfrag[ni].q[0] = *(const uint4*)(&lb[col * 32 + kb]);
            bfrag[ni].q[1] = *(const uint4*)(&lb[col * 32 + kb + 16]);
        }
#pragma unroll
        for (int mi = 0; mi < 2; ++mi)
#pragma unroll
            for (int ni = 0; ni < 4; ++ni)
                acc[mi][ni] = __builtin_amdgcn_wmma_f32_16x16x32_bf16(
                    false, afrag[mi].v, false, bfrag[ni].v,
                    (short)0, acc[mi][ni], false, false);

        // ---- commit next tile into the other buffer (no barrier needed:
        //      writes go to buf[p^1], reads above were from buf[p]) ----
        if (hasNext) commit(p ^ 1);
        p ^= 1;
    }

    // ---- epilogue: bias add + store ----
#pragma unroll
    for (int mi = 0; mi < 2; ++mi) {
#pragma unroll
        for (int ni = 0; ni < 4; ++ni) {
            int col  = nBlock + wc * 64 + ni * 16 + lane15;
            float bv = bias[col];
#pragma unroll
            for (int r = 0; r < 8; ++r) {
                int row = mBlock + wr * 32 + mi * 16 + r + half * 8;
                C[(size_t)row * ldc + col] = acc[mi][ni][r] + bv;
            }
        }
    }
}

// ---------------------------------------------------------------------------
// Flash-style causal attention for one depth, with spiral gate blend.
// One wave per 16 query rows; 8 waves (128 q rows) per workgroup.
// q/k/v live inside the qkv buffer (B,T,3C); att is merged (B,T,C) layout.
// ---------------------------------------------------------------------------
__global__ __launch_bounds__(256) void spiral_attn(
    const float* __restrict__ qkv, float* __restrict__ att,
    const float* __restrict__ gates, int depth)
{
    const int T = 2048, C = 1024, C3 = 3072, Dh = 64;
    __shared__ __align__(16) __bf16 lds_p[8][16 * 64];

    const int tid    = threadIdx.x;
    const int lane   = tid & 31;
    const int wave   = tid >> 5;
    const int lane15 = lane & 15;
    const int half   = lane >> 4;
    const int b      = blockIdx.z;
    const int h      = blockIdx.y;
    const int qb0    = blockIdx.x * 128 + wave * 16;
    const float inv_sqrt_d = 0.125f;

    const size_t baseQ = (size_t)b * T * C3 + (size_t)h * Dh;
    const size_t baseK = baseQ + C;
    const size_t baseV = baseQ + 2 * C;

    union FragU { v16bf v; bf16x2 p[8]; uint4 q[2]; };

    // ---- load Q fragments once (A-matrix layout, D=64 -> 2 K-steps) ----
    FragU aq[2];
    {
        const float* qrow = qkv + baseQ + (size_t)(qb0 + lane15) * C3;
#pragma unroll
        for (int ks = 0; ks < 2; ++ks) {
            int d0 = ks * 32 + half * 8;
#pragma unroll
            for (int e = 0; e < 4; ++e)
                aq[ks].p[e]     = cvt_pk_bf16(qrow[d0 + 2 * e],      qrow[d0 + 2 * e + 1]);
#pragma unroll
            for (int e = 0; e < 4; ++e)
                aq[ks].p[4 + e] = cvt_pk_bf16(qrow[d0 + 16 + 2 * e], qrow[d0 + 17 + 2 * e]);
        }
    }

    float mrun[8], lrun[8];
    v8f accO[4] = {};
#pragma unroll
    for (int r = 0; r < 8; ++r) { mrun[r] = -3.0e38f; lrun[r] = 0.0f; }

    const int jEnd = qb0 >> 6;                 // wave-uniform causal bound
    for (int j = 0; j <= jEnd; ++j) {
        const int jbase = j << 6;
        const bool diag = (j == jEnd);

        // ---- S = Q K^T (K fragments from global: [t][d] matches B layout) ----
        v8f s[4] = {};
#pragma unroll
        for (int ni = 0; ni < 4; ++ni) {
            const float* krow = qkv + baseK + (size_t)(jbase + ni * 16 + lane15) * C3;
#pragma unroll
            for (int ks = 0; ks < 2; ++ks) {
                FragU bk;
                int d0 = ks * 32 + half * 8;
#pragma unroll
                for (int e = 0; e < 4; ++e)
                    bk.p[e]     = cvt_pk_bf16(krow[d0 + 2 * e],      krow[d0 + 2 * e + 1]);
#pragma unroll
                for (int e = 0; e < 4; ++e)
                    bk.p[4 + e] = cvt_pk_bf16(krow[d0 + 16 + 2 * e], krow[d0 + 17 + 2 * e]);
                s[ni] = __builtin_amdgcn_wmma_f32_16x16x32_bf16(
                    false, aq[ks].v, false, bk.v, (short)0, s[ni], false, false);
            }
        }

        // ---- scale + causal mask (lane-select, EXEC stays all-ones) ----
#pragma unroll
        for (int ni = 0; ni < 4; ++ni) {
            int kvc = jbase + ni * 16 + lane15;
#pragma unroll
            for (int r = 0; r < 8; ++r) {
                float v = s[ni][r] * inv_sqrt_d;
                if (diag) {
                    int qr = qb0 + r + half * 8;
                    v = (kvc <= qr) ? v : -3.0e38f;
                }
                s[ni][r] = v;
            }
        }

        // ---- online softmax: rowmax / exp / rowsum via 16-lane butterflies ----
        float mnew[8], rsum[8];
#pragma unroll
        for (int r = 0; r < 8; ++r) {
            float m = fmaxf(fmaxf(s[0][r], s[1][r]), fmaxf(s[2][r], s[3][r]));
#pragma unroll
            for (int off = 1; off < 16; off <<= 1) m = fmaxf(m, __shfl_xor(m, off, 32));
            mnew[r] = fmaxf(mrun[r], m);
        }
#pragma unroll
        for (int r = 0; r < 8; ++r) {
            float sum = 0.f;
#pragma unroll
            for (int ni = 0; ni < 4; ++ni) {
                float p = __expf(s[ni][r] - mnew[r]);
                s[ni][r] = p;
                sum += p;
            }
#pragma unroll
            for (int off = 1; off < 16; off <<= 1) sum += __shfl_xor(sum, off, 32);
            rsum[r] = sum;
        }
#pragma unroll
        for (int r = 0; r < 8; ++r) {
            float alpha = __expf(mrun[r] - mnew[r]);
            lrun[r] = lrun[r] * alpha + rsum[r];
            mrun[r] = mnew[r];
#pragma unroll
            for (int ni = 0; ni < 4; ++ni) accO[ni][r] *= alpha;
        }

        // ---- stage P (bf16) to per-wave LDS, reload as A fragments ----
        __bf16* pbuf = lds_p[wave];
#pragma unroll
        for (int ni = 0; ni < 4; ++ni)
#pragma unroll
            for (int r = 0; r < 8; ++r)
                pbuf[(r + half * 8) * 64 + ni * 16 + lane15] = (__bf16)s[ni][r];

#pragma unroll
        for (int ks = 0; ks < 2; ++ks) {
            FragU ap;
            int kb = ks * 32 + half * 8;
            ap.q[0] = *(const uint4*)(&pbuf[lane15 * 64 + kb]);
            ap.q[1] = *(const uint4*)(&pbuf[lane15 * 64 + kb + 16]);
#pragma unroll
            for (int ni = 0; ni < 4; ++ni) {
                FragU bv;
                const float* vcol = qkv + baseV + ni * 16 + lane15;
#pragma unroll
                for (int e = 0; e < 4; ++e)
                    bv.p[e]     = cvt_pk_bf16(vcol[(size_t)(jbase + kb + 2 * e) * C3],
                                              vcol[(size_t)(jbase + kb + 2 * e + 1) * C3]);
#pragma unroll
                for (int e = 0; e < 4; ++e)
                    bv.p[4 + e] = cvt_pk_bf16(vcol[(size_t)(jbase + kb + 16 + 2 * e) * C3],
                                              vcol[(size_t)(jbase + kb + 17 + 2 * e) * C3]);
                accO[ni] = __builtin_amdgcn_wmma_f32_16x16x32_bf16(
                    false, ap.v, false, bv.v, (short)0, accO[ni], false, false);
            }
        }
    }

    // ---- finalize, spiral gate blend, store merged (B,T,C) ----
    float g = 1.0f / (1.0f + __expf(-gates[depth]));
#pragma unroll
    for (int ni = 0; ni < 4; ++ni) {
        int d = ni * 16 + lane15;
#pragma unroll
        for (int r = 0; r < 8; ++r) {
            int t = qb0 + r + half * 8;
            size_t idx = ((size_t)b * T + t) * C + (size_t)h * Dh + d;
            float o = accO[ni][r] / lrun[r];
            att[idx] = (depth == 0) ? o : (g * o + (1.0f - g) * att[idx]);
        }
    }
}

// ---------------------------------------------------------------------------
extern "C" void kernel_launch(void* const* d_in, const int* in_sizes, int n_in,
                              void* d_out, int out_size, void* d_ws, size_t ws_size,
                              hipStream_t stream)
{
    (void)in_sizes; (void)n_in; (void)out_size; (void)ws_size;

    const float* x      = (const float*)d_in[0];
    const float* W_attn = (const float*)d_in[1];
    const float* b_attn = (const float*)d_in[2];
    const float* W_proj = (const float*)d_in[3];
    const float* b_proj = (const float*)d_in[4];
    const float* gates  = (const float*)d_in[5];
    float*       y      = (float*)d_out;

    const int B = 2, T = 2048, C = 1024, H = 16;
    const int M = B * T;                       // 4096

    float* qkv = (float*)d_ws;                 // M * 3C floats (48 MB)
    float* att = qkv + (size_t)M * 3 * C;      // M * C  floats (16 MB)

    dim3 blk(256);

    // qkv = x @ W_attn + b_attn
    gemm_bias_bf16<<<dim3(3 * C / 128, M / 128), blk, 0, stream>>>(
        x, W_attn, b_attn, qkv, M, 3 * C, C, C, 3 * C, 3 * C);

    for (int depth = 0; depth < 3; ++depth) {
        spiral_attn<<<dim3(T / 128, H, B), blk, 0, stream>>>(qkv, att, gates, depth);
        if (depth < 2) {
            // refine q,k: att @ W_attn[:, :2C] + b_attn[:2C] -> qkv cols [0,2C)
            gemm_bias_bf16<<<dim3(2 * C / 128, M / 128), blk, 0, stream>>>(
                att, W_attn, b_attn, qkv, M, 2 * C, C, C, 3 * C, 3 * C);
        }
    }

    // y = att @ W_proj + b_proj
    gemm_bias_bf16<<<dim3(C / 128, M / 128), blk, 0, stream>>>(
        att, W_proj, b_proj, y, M, C, C, C, C, C);
}
